// RefineVAE_33887291965773
// MI455X (gfx1250) — matrix operand order repaired and checked by reading, not compiled, not run on hardware
//
#include <hip/hip_runtime.h>
#include <math.h>

// ---------------------------------------------------------------------------
// RefineVAE forward for MI455X (gfx1250, wave32, WMMA).
// GEMMs: v_wmma_f32_16x16x32_bf16. fp32->bf16 via single v_perm_b32 per pair.
// Double-buffered LDS with register staging: next tile's global_load_b128s
// are issued before the WMMA compute of the current tile.
// A staged row-major (frag = 2x ds_load_b128), B staged TRANSPOSED so each
// lane's fragment is two 16-byte runs, k=8h..8h+7 and 16+8h..23+8h (2x ds_load_b128).
// Attention is computed windowed (|j-i|<=15) instead of dense TxT.
// ---------------------------------------------------------------------------

typedef __bf16 bf16_t;
typedef __attribute__((ext_vector_type(16))) __bf16 v16bf;
typedef __attribute__((ext_vector_type(8)))  __bf16 v8bf;
typedef __attribute__((ext_vector_type(8)))  float  v8f;

#define DMODEL  512
#define NHEADS  8
#define DHEAD   64
#define DFF     2048
#define NLAYERS 6
#define HALFW   15
#define WIN     31          // 2*HALFW+1
#define CTXF    10
#define DMOT    147
#define BSZ     32
#define TT      240         // frames
#define TE      242         // frames + 2 tokens (encoder)
#define ATT_SCALE 0.125f    // 1/sqrt(64)

#define BM 128
#define BN 64
#define BK 32
#define BKP 40              // padded LDS row stride (halves); 80B, 16B-aligned

__device__ __forceinline__ int iclamp(int v, int hi) { return v < hi ? v : hi; }

// two fp32 -> packed bf16x2 (truncation) in ONE v_perm_b32
__device__ __forceinline__ unsigned pack2bf(float lo, float hi) {
  return __builtin_amdgcn_perm(__float_as_uint(hi), __float_as_uint(lo), 0x07060302u);
}

// ---------------------------------------------------------------------------
// Tiled GEMM: C[M,N] = act( A[M,K] @ W[K,N] + bias + resid )
// 256 threads = 8 waves -> 128x64 tile; each wave owns 32x32 (2x2 WMMA).
// act: 0=none, 1=relu, 2=prelu(alpha).
// ---------------------------------------------------------------------------
__global__ __launch_bounds__(256, 2) void k_gemm(
    const float* __restrict__ A, int lda,
    const float* __restrict__ W, int ldw,
    const float* __restrict__ bias,
    const float* __restrict__ resid,
    float* __restrict__ C, int ldc,
    int M, int N, int K, int act, const float* __restrict__ alpha_p)
{
  __shared__ alignas(16) bf16_t As[2][BM][BKP];   // row-major   (2x 10.2 KB)
  __shared__ alignas(16) bf16_t BsT[2][BN][BKP];  // transposed  (2x  5.1 KB)

  const int tid  = threadIdx.x;
  const int lane = tid & 31;
  const int wave = tid >> 5;
  const int wm = wave & 3;   // 4 waves along M
  const int wn = wave >> 2;  // 2 waves along N
  const int hi  = lane >> 4;
  const int m16 = lane & 15;

  const int rowBase = blockIdx.y * BM;
  const int colBase = blockIdx.x * BN;

  const bool fastA0 = (rowBase + BM <= M) && ((lda & 3) == 0);
  const bool fastB0 = (colBase + BN <= N) && ((ldw & 3) == 0);

  // staging layout: A -> thread = (row, 16-wide K chunk); B -> 2(K) x 4(N)
  const int ar_ = tid >> 1;            // 0..127
  const int ak0 = (tid & 1) * 16;      // 0 | 16
  const int bk_ = (tid >> 4) * 2;      // 0..30 (even)
  const int bc_ = (tid & 15) * 4;      // 0..60

  float av[16], bv[8];

  auto gloadA = [&](int k0) {
    if (fastA0 && (k0 + BK <= K)) {
      const float4* p = (const float4*)(A + (size_t)(rowBase + ar_) * lda + k0 + ak0);
      float4 f0 = p[0], f1 = p[1], f2 = p[2], f3 = p[3];
      av[0] = f0.x; av[1] = f0.y; av[2]  = f0.z; av[3]  = f0.w;
      av[4] = f1.x; av[5] = f1.y; av[6]  = f1.z; av[7]  = f1.w;
      av[8] = f2.x; av[9] = f2.y; av[10] = f2.z; av[11] = f2.w;
      av[12] = f3.x; av[13] = f3.y; av[14] = f3.z; av[15] = f3.w;
    } else {
      const int gr = rowBase + ar_;
      const size_t rb = (size_t)iclamp(gr, M - 1) * lda;
#pragma unroll
      for (int i = 0; i < 16; ++i) {
        const int gk = k0 + ak0 + i;
        float v = A[rb + iclamp(gk, K - 1)];
        av[i] = (gr < M && gk < K) ? v : 0.f;
      }
    }
  };
  auto gloadB = [&](int k0) {
    if (fastB0 && (k0 + BK <= K)) {
      const float* base = W + (size_t)(k0 + bk_) * ldw + colBase + bc_;
      float4 w0 = *(const float4*)base;
      float4 w1 = *(const float4*)(base + ldw);
      bv[0] = w0.x; bv[1] = w0.y; bv[2] = w0.z; bv[3] = w0.w;
      bv[4] = w1.x; bv[5] = w1.y; bv[6] = w1.z; bv[7] = w1.w;
    } else {
#pragma unroll
      for (int i = 0; i < 8; ++i) {
        const int gk = k0 + bk_ + (i >> 2);
        const int gc = colBase + bc_ + (i & 3);
        float v = W[(size_t)iclamp(gk, K - 1) * ldw + iclamp(gc, N - 1)];
        bv[i] = (gk < K && gc < N) ? v : 0.f;
      }
    }
  };
  auto stage = [&](int buf) {
    unsigned u[8];
#pragma unroll
    for (int i = 0; i < 8; ++i) u[i] = pack2bf(av[2 * i], av[2 * i + 1]);
    uint4 alo; alo.x = u[0]; alo.y = u[1]; alo.z = u[2]; alo.w = u[3];
    uint4 ahi; ahi.x = u[4]; ahi.y = u[5]; ahi.z = u[6]; ahi.w = u[7];
    *(uint4*)&As[buf][ar_][ak0]     = alo;
    *(uint4*)&As[buf][ar_][ak0 + 8] = ahi;
#pragma unroll
    for (int i = 0; i < 4; ++i)                       // K-pair -> one b32 store
      *(unsigned*)&BsT[buf][bc_ + i][bk_] = pack2bf(bv[i], bv[4 + i]);
  };

  v8f acc[2][2];
#pragma unroll
  for (int i = 0; i < 2; ++i)
#pragma unroll
    for (int j = 0; j < 2; ++j)
#pragma unroll
      for (int r = 0; r < 8; ++r) acc[i][j][r] = 0.f;

  gloadA(0);
  gloadB(0);
  int buf = 0;
  for (int k0 = 0; k0 < K; k0 += BK, buf ^= 1) {
    stage(buf);
    __syncthreads();
    if (k0 + BK < K) { gloadA(k0 + BK); gloadB(k0 + BK); }  // overlap with WMMA

    v16bf bfr[2];
#pragma unroll
    for (int j = 0; j < 2; ++j) {
      const int bc = wn * 32 + j * 16 + m16;
      v8bf b0 = *(const v8bf*)&BsT[buf][bc][hi * 8];
      v8bf b1 = *(const v8bf*)&BsT[buf][bc][16 + hi * 8];
      bfr[j] = __builtin_shufflevector(b0, b1, 0, 1, 2, 3, 4, 5, 6, 7,
                                       8, 9, 10, 11, 12, 13, 14, 15);
    }
    v16bf af[2];
#pragma unroll
    for (int i = 0; i < 2; ++i) {
      const int ar = wm * 32 + i * 16 + m16;
      v8bf a0 = *(const v8bf*)&As[buf][ar][hi * 8];
      v8bf a1 = *(const v8bf*)&As[buf][ar][16 + hi * 8];
      af[i] = __builtin_shufflevector(a0, a1, 0, 1, 2, 3, 4, 5, 6, 7,
                                      8, 9, 10, 11, 12, 13, 14, 15);
    }
#pragma unroll
    for (int i = 0; i < 2; ++i)
#pragma unroll
      for (int j = 0; j < 2; ++j)
        acc[i][j] = __builtin_amdgcn_wmma_f32_16x16x32_bf16(
            false, af[i], false, bfr[j], (short)0, acc[i][j], false, false);
  }

  const float alpha = (act == 2) ? alpha_p[0] : 0.f;
#pragma unroll
  for (int i = 0; i < 2; ++i)
#pragma unroll
    for (int j = 0; j < 2; ++j)
#pragma unroll
      for (int r = 0; r < 8; ++r) {
        const int row = rowBase + wm * 32 + i * 16 + (r + 8 * hi);
        const int col = colBase + wn * 32 + j * 16 + m16;
        if (row < M && col < N) {
          float v = acc[i][j][r];
          if (bias)  v += bias[col];
          if (resid) v += resid[(size_t)row * ldc + col];
          if (act == 1)      v = fmaxf(v, 0.f);
          else if (act == 2) v = (v >= 0.f) ? v : alpha * v;
          C[(size_t)row * ldc + col] = v;
        }
      }
}

// ---------------------------------------------------------------------------
// Row LayerNorm over 512 features, one wave per row (16 elems/lane).
// ---------------------------------------------------------------------------
__global__ __launch_bounds__(32) void k_ln(
    const float* __restrict__ X, const float* __restrict__ g,
    const float* __restrict__ b, float* __restrict__ Y, int nrows)
{
  const int row = blockIdx.x;
  if (row >= nrows) return;
  const int lane = threadIdx.x;
  const float* x = X + (size_t)row * DMODEL;
  float v[16], s = 0.f, ss = 0.f;
#pragma unroll
  for (int i = 0; i < 16; ++i) {
    float t = x[lane + i * 32];
    v[i] = t; s += t; ss += t * t;
  }
#pragma unroll
  for (int o = 16; o; o >>= 1) { s += __shfl_xor(s, o, 32); ss += __shfl_xor(ss, o, 32); }
  const float mean = s * (1.f / DMODEL);
  const float var  = ss * (1.f / DMODEL) - mean * mean;
  const float inv  = rsqrtf(var + 1e-5f);
  float* y = Y + (size_t)row * DMODEL;
#pragma unroll
  for (int i = 0; i < 16; ++i) {
    const int d = lane + i * 32;
    y[d] = (v[i] - mean) * inv * g[d] + b[d];
  }
}

// ---------------------------------------------------------------------------
// Windowed attention with relative-position bias:
// scores_ij = (q_i . (k_j + E[j-i+15])) * SCALE for |j-i|<=15.
// Grid (t, B), block 256 = 8 waves = 8 heads; 2 d_head elems per lane.
// ---------------------------------------------------------------------------
__global__ __launch_bounds__(256) void k_attn(
    const float* __restrict__ Q, const float* __restrict__ Kv,
    const float* __restrict__ Vv, const float* __restrict__ lut,
    float* __restrict__ O, int t)
{
  const int i = blockIdx.x, b = blockIdx.y;
  const int h = threadIdx.x >> 5, lane = threadIdx.x & 31;
  const size_t qoff = (size_t)(b * t + i) * DMODEL + h * DHEAD + lane * 2;
  const float q0 = Q[qoff], q1 = Q[qoff + 1];

  float s[WIN];
#pragma unroll
  for (int w = 0; w < WIN; ++w) {
    const int j = i - HALFW + w;
    float sc = -1e30f;
    if (j >= 0 && j < t) {
      const float e0 = lut[w * DHEAD + lane * 2];
      const float e1 = lut[w * DHEAD + lane * 2 + 1];
      const size_t ko = (size_t)(b * t + j) * DMODEL + h * DHEAD + lane * 2;
      float p = q0 * (Kv[ko] + e0) + q1 * (Kv[ko + 1] + e1);
#pragma unroll
      for (int o = 16; o; o >>= 1) p += __shfl_xor(p, o, 32);
      sc = p * ATT_SCALE;
    }
    s[w] = sc;
  }
  float m = -1e30f;
#pragma unroll
  for (int w = 0; w < WIN; ++w) m = fmaxf(m, s[w]);
  float sum = 0.f;
#pragma unroll
  for (int w = 0; w < WIN; ++w) {
    const float e = (s[w] > -1e29f) ? __expf(s[w] - m) : 0.f;
    s[w] = e; sum += e;
  }
  const float inv = 1.f / sum;
  float a0 = 0.f, a1 = 0.f;
#pragma unroll
  for (int w = 0; w < WIN; ++w) {
    const int j = i - HALFW + w;
    if (j >= 0 && j < t && s[w] != 0.f) {
      const size_t vo = (size_t)(b * t + j) * DMODEL + h * DHEAD + lane * 2;
      a0 += s[w] * Vv[vo];
      a1 += s[w] * Vv[vo + 1];
    }
  }
  O[qoff]     = a0 * inv;
  O[qoff + 1] = a1 * inv;
}

// ---------------------------------------------------------------------------
// Relative-position lookup table core (31 x 64): prelu(rel@W1+b1,a)@W2+b2.
// ---------------------------------------------------------------------------
__global__ __launch_bounds__(256) void k_lut(
    const float* __restrict__ W1, const float* __restrict__ b1,
    const float* __restrict__ a_p, const float* __restrict__ W2,
    const float* __restrict__ b2, float* __restrict__ lut)
{
  __shared__ float h[WIN * DMODEL];   // 63.5 KB (LDS = 320 KB on CDNA5)
  const float a = a_p[0];
  for (int idx = threadIdx.x; idx < WIN * DMODEL; idx += 256) {
    const int r = idx >> 9, c = idx & 511;
    const float rel = (float)(r - HALFW);
    const float v = rel * W1[c] + b1[c];
    h[idx] = (v >= 0.f) ? v : a * v;
  }
  __syncthreads();
  for (int idx = threadIdx.x; idx < WIN * DHEAD; idx += 256) {
    const int r = idx >> 6, d = idx & 63;
    float acc = b2[d];
    const float* hr = &h[r * DMODEL];
    for (int c = 0; c < DMODEL; ++c) acc += hr[c] * W2[c * DHEAD + d];
    lut[idx] = acc;
  }
}

// ---------------------------------------------------------------------------
// Small glue kernels.
// ---------------------------------------------------------------------------
__global__ void k_assemble_enc(const float* __restrict__ mlp,
                               const float* __restrict__ mtok,
                               const float* __restrict__ lvtok,
                               float* __restrict__ H)
{
  const int idx = blockIdx.x * 256 + threadIdx.x;
  if (idx >= BSZ * TE * DMODEL) return;
  const int d = idx & 511;
  const int r = (idx >> 9) % TE;
  const int b = idx / (TE * DMODEL);
  float v;
  if (r == 0)      v = mtok[d];
  else if (r == 1) v = lvtok[d];
  else             v = mlp[(size_t)(b * TT + (r - 2)) * DMODEL + d];
  H[idx] = v;
}

__global__ void k_extract(const float* __restrict__ Hln,
                          float* __restrict__ mean_ws, float* __restrict__ lv_ws,
                          float* __restrict__ mean_out, float* __restrict__ lv_out)
{
  const int idx = blockIdx.x * 256 + threadIdx.x;
  if (idx >= BSZ * DMODEL) return;
  const int b = idx >> 9, d = idx & 511;
  const float m  = Hln[(size_t)(b * TE) * DMODEL + d];
  const float lv = Hln[(size_t)(b * TE + 1) * DMODEL + d];
  mean_ws[idx] = m;  lv_ws[idx] = lv;
  mean_out[idx] = m; lv_out[idx] = lv;
}

__global__ void k_reparam(const float* __restrict__ mean, const float* __restrict__ lv,
                          const float* __restrict__ noise, float* __restrict__ z)
{
  const int idx = blockIdx.x * 256 + threadIdx.x;
  if (idx >= BSZ * TT * DMODEL) return;
  const int d = idx & 511;
  const int b = idx / (TT * DMODEL);
  z[idx] = mean[b * DMODEL + d] + noise[idx] * __expf(0.5f * lv[b * DMODEL + d]);
}

__global__ void k_concat_dec(const float* __restrict__ xi, const int* __restrict__ kf,
                             float* __restrict__ H)
{
  const int DIN = DMOT + 1;
  const int idx = blockIdx.x * 256 + threadIdx.x;
  if (idx >= BSZ * TT * DIN) return;
  const int c = idx % DIN;
  const int t = (idx / DIN) % TT;
  const int b = idx / (DIN * TT);
  float v;
  if (c < DMOT) {
    v = xi[(size_t)(b * TT + t) * DMOT + c];
  } else {
    int m = (t < CTXF) || (t == TT - 1);
#pragma unroll
    for (int q = 0; q < 6; ++q) m |= (t == kf[q]);
    v = m ? 1.f : 0.f;
  }
  H[idx] = v;
}

// ---------------------------------------------------------------------------
// Host-side orchestration.
// Input flattening (setup_inputs() insertion order, nested dicts flattened in
// insertion order): 0:x 1:x_interp 2:noise 3:keyframes, 4..34: enc leaves,
// 35..68: dec leaves.
// ---------------------------------------------------------------------------
enum { P_meW1=0, P_meb1, P_mea1, P_meW2, P_meb2, P_mea2,
       P_rpW1, P_rpb1, P_rpa, P_rpW2, P_rpb2,
       P_Wq, P_bq, P_Wk, P_bk, P_Wv, P_bv, P_Wo, P_bo,
       P_atg, P_atb, P_ffW1, P_ffb1, P_ffW2, P_ffb2, P_ffg, P_ffb,
       P_lng, P_lnb, P_mtok, P_lvtok,
       P_hdW1 = 29, P_hdb1 = 30, P_hda = 31, P_hdW2 = 32, P_hdb2 = 33 };

extern "C" void kernel_launch(void* const* d_in, const int* in_sizes, int n_in,
                              void* d_out, int out_size, void* d_ws, size_t ws_size,
                              hipStream_t stream)
{
  (void)in_sizes; (void)n_in; (void)out_size; (void)ws_size;

  const float* x     = (const float*)d_in[0];
  const float* xi    = (const float*)d_in[1];
  const float* noise = (const float*)d_in[2];
  const int*   kf    = (const int*)d_in[3];
  const float* const* Ep = (const float* const*)&d_in[4];
  const float* const* Dp = (const float* const*)&d_in[35];

  const int ME = BSZ * TE;   // 7744 encoder rows
  const int MT = BSZ * TT;   // 7680 frame rows
  const int DOUT = DMOT - 3; // 144

  // workspace carve-out (floats)
  float* ws = (float*)d_ws;
  size_t off = 0;
  auto alloc = [&](size_t n) { float* p = ws + off; off += n; return p; };
  float* lutE = alloc(WIN * DHEAD);
  float* lutD = alloc(WIN * DHEAD);
  float* H    = alloc((size_t)ME * DMODEL);
  float* LN1  = alloc((size_t)ME * DMODEL);
  float* Qb   = alloc((size_t)ME * DMODEL);
  float* Kb   = alloc((size_t)ME * DMODEL);
  float* Vb   = alloc((size_t)ME * DMODEL);
  float* Ab   = alloc((size_t)ME * DMODEL);
  float* FF   = alloc((size_t)ME * DFF);
  float* M1   = alloc((size_t)ME * DMODEL);
  float* Z    = alloc((size_t)MT * DMODEL);
  float* LKV  = alloc((size_t)MT * DMODEL);
  float* HIN  = alloc((size_t)MT * (DMOT + 1));
  float* MEAN = alloc((size_t)BSZ * DMODEL);
  float* LV   = alloc((size_t)BSZ * DMODEL);

  auto gemm = [&](const float* A, int lda, const float* W, int ldw,
                  const float* bias, const float* res, float* C, int ldc,
                  int M, int N, int K, int act, const float* al) {
    dim3 gg((N + BN - 1) / BN, (M + BM - 1) / BM);
    k_gemm<<<gg, 256, 0, stream>>>(A, lda, W, ldw, bias, res, C, ldc, M, N, K, act, al);
  };
  auto ln = [&](const float* X, const float* gg, const float* bb, float* Y, int rows) {
    k_ln<<<rows, 32, 0, stream>>>(X, gg, bb, Y, rows);
  };

  // relative-position LUTs (enc / dec have separate params)
  k_lut<<<1, 256, 0, stream>>>(Ep[P_rpW1], Ep[P_rpb1], Ep[P_rpa], Ep[P_rpW2], Ep[P_rpb2], lutE);
  k_lut<<<1, 256, 0, stream>>>(Dp[P_rpW1], Dp[P_rpb1], Dp[P_rpa], Dp[P_rpW2], Dp[P_rpb2], lutD);

  // ===================== encoder =====================
  gemm(x, DMOT, Ep[P_meW1], DMODEL, Ep[P_meb1], nullptr, M1, DMODEL, MT, DMODEL, DMOT, 2, Ep[P_mea1]);
  gemm(M1, DMODEL, Ep[P_meW2], DMODEL, Ep[P_meb2], nullptr, Ab, DMODEL, MT, DMODEL, DMODEL, 2, Ep[P_mea2]);
  k_assemble_enc<<<(ME * DMODEL + 255) / 256, 256, 0, stream>>>(Ab, Ep[P_mtok], Ep[P_lvtok], H);

  for (int l = 0; l < NLAYERS; ++l) {
    const size_t wofs = (size_t)l * DMODEL * DMODEL, bofs = (size_t)l * DMODEL;
    ln(H, Ep[P_atg] + bofs, Ep[P_atb] + bofs, LN1, ME);
    gemm(LN1, DMODEL, Ep[P_Wq] + wofs, DMODEL, Ep[P_bq] + bofs, nullptr, Qb, DMODEL, ME, DMODEL, DMODEL, 0, nullptr);
    gemm(LN1, DMODEL, Ep[P_Wk] + wofs, DMODEL, Ep[P_bk] + bofs, nullptr, Kb, DMODEL, ME, DMODEL, DMODEL, 0, nullptr);
    gemm(LN1, DMODEL, Ep[P_Wv] + wofs, DMODEL, Ep[P_bv] + bofs, nullptr, Vb, DMODEL, ME, DMODEL, DMODEL, 0, nullptr);
    k_attn<<<dim3(TE, BSZ), 256, 0, stream>>>(Qb, Kb, Vb, lutE, Ab, TE);
    gemm(Ab, DMODEL, Ep[P_Wo] + wofs, DMODEL, Ep[P_bo] + bofs, H, H, DMODEL, ME, DMODEL, DMODEL, 0, nullptr);
    ln(H, Ep[P_ffg] + bofs, Ep[P_ffb] + bofs, LN1, ME);
    gemm(LN1, DMODEL, Ep[P_ffW1] + (size_t)l * DMODEL * DFF, DFF, Ep[P_ffb1] + (size_t)l * DFF,
         nullptr, FF, DFF, ME, DFF, DMODEL, 1, nullptr);
    gemm(FF, DFF, Ep[P_ffW2] + (size_t)l * DFF * DMODEL, DMODEL, Ep[P_ffb2] + bofs, H, H, DMODEL, ME, DMODEL, DFF, 0, nullptr);
  }
  ln(H, Ep[P_lng], Ep[P_lnb], LN1, ME);

  float* out_recon = (float*)d_out;
  float* out_mean  = out_recon + (size_t)MT * DOUT;
  float* out_lv    = out_mean + (size_t)BSZ * DMODEL;
  k_extract<<<(BSZ * DMODEL + 255) / 256, 256, 0, stream>>>(LN1, MEAN, LV, out_mean, out_lv);
  k_reparam<<<(MT * DMODEL + 255) / 256, 256, 0, stream>>>(MEAN, LV, noise, Z);

  // ===================== decoder =====================
  k_concat_dec<<<(MT * (DMOT + 1) + 255) / 256, 256, 0, stream>>>(xi, kf, HIN);
  gemm(HIN, DMOT + 1, Dp[P_meW1], DMODEL, Dp[P_meb1], nullptr, M1, DMODEL, MT, DMODEL, DMOT + 1, 2, Dp[P_mea1]);
  gemm(M1, DMODEL, Dp[P_meW2], DMODEL, Dp[P_meb2], nullptr, H, DMODEL, MT, DMODEL, DMODEL, 2, Dp[P_mea2]);

  for (int l = 0; l < NLAYERS; ++l) {
    const size_t wofs = (size_t)l * DMODEL * DMODEL, bofs = (size_t)l * DMODEL;
    ln(H, Dp[P_atg] + bofs, Dp[P_atb] + bofs, LN1, MT);
    ln(Z, Dp[P_atg] + bofs, Dp[P_atb] + bofs, LKV, MT);   // cross-attend to latent z
    gemm(LN1, DMODEL, Dp[P_Wq] + wofs, DMODEL, Dp[P_bq] + bofs, nullptr, Qb, DMODEL, MT, DMODEL, DMODEL, 0, nullptr);
    gemm(LKV, DMODEL, Dp[P_Wk] + wofs, DMODEL, Dp[P_bk] + bofs, nullptr, Kb, DMODEL, MT, DMODEL, DMODEL, 0, nullptr);
    gemm(LKV, DMODEL, Dp[P_Wv] + wofs, DMODEL, Dp[P_bv] + bofs, nullptr, Vb, DMODEL, MT, DMODEL, DMODEL, 0, nullptr);
    k_attn<<<dim3(TT, BSZ), 256, 0, stream>>>(Qb, Kb, Vb, lutD, Ab, TT);
    gemm(Ab, DMODEL, Dp[P_Wo] + wofs, DMODEL, Dp[P_bo] + bofs, H, H, DMODEL, MT, DMODEL, DMODEL, 0, nullptr);
    ln(H, Dp[P_ffg] + bofs, Dp[P_ffb] + bofs, LN1, MT);
    gemm(LN1, DMODEL, Dp[P_ffW1] + (size_t)l * DMODEL * DFF, DFF, Dp[P_ffb1] + (size_t)l * DFF,
         nullptr, FF, DFF, MT, DFF, DMODEL, 1, nullptr);
    gemm(FF, DFF, Dp[P_ffW2] + (size_t)l * DFF * DMODEL, DMODEL, Dp[P_ffb2] + bofs, H, H, DMODEL, MT, DMODEL, DFF, 0, nullptr);
  }
  ln(H, Dp[P_lng], Dp[P_lnb], LN1, MT);
  gemm(LN1, DMODEL, Dp[P_hdW1], DMODEL, Dp[P_hdb1], nullptr, M1, DMODEL, MT, DMODEL, DMODEL, 2, Dp[P_hda]);
  gemm(M1, DMODEL, Dp[P_hdW2], DOUT, Dp[P_hdb2], nullptr, out_recon, DOUT, MT, DOUT, DMODEL, 0, nullptr);
}
